// MambaModel_2233382994679
// MI455X (gfx1250) — matrix operand order repaired and checked by reading, not compiled, not run on hardware
//
#include <hip/hip_runtime.h>
#include <cstddef>

// ---------------- model dimensions ----------------
#define NB      4
#define LSEQ    1024
#define BLROWS  (NB * LSEQ)        // 4096
#define DMODEL  512
#define DINNER  1024
#define HEADDIM 64
#define NHEADS  16
#define DSTATE  64
#define DCONV   4
#define CONVDIM 1152               // DINNER + 2*DSTATE
#define DINPROJ 2192               // 2*DINNER + 2*DSTATE + NHEADS
#define NCLSOUT 164
#define LN_EPS  1e-5f
#define MAXVAL  512.0f

typedef __attribute__((ext_vector_type(16))) _Float16 v16h;
typedef __attribute__((ext_vector_type(8)))  _Float16 v8h;
typedef __attribute__((ext_vector_type(8)))  float    v8f;

__device__ __forceinline__ float siluf(float x)    { return x / (1.0f + expf(-x)); }
__device__ __forceinline__ float softplusf(float x){ return fmaxf(x, 0.0f) + log1pf(expf(-fabsf(x))); }

// =====================================================================
// f32 -> f16 streaming conversion (activations & weights)
// =====================================================================
__global__ __launch_bounds__(256)
void f32_to_f16_kernel(const float* __restrict__ in, _Float16* __restrict__ out, int n)
{
    int i0 = (blockIdx.x * 256 + threadIdx.x) * 4;
    if (i0 + 3 < n) {
        float4 v = *(const float4*)(in + i0);
        out[i0 + 0] = (_Float16)v.x;
        out[i0 + 1] = (_Float16)v.y;
        out[i0 + 2] = (_Float16)v.z;
        out[i0 + 3] = (_Float16)v.w;
    } else {
        for (int i = i0; i < n; ++i) out[i] = (_Float16)in[i];
    }
}

// =====================================================================
// WMMA GEMM:  C[M,N] = act(A[M,K] @ W[N,K]^T + bias[N]),  A/W in f16.
// Each wave computes a 32x32 tile: 2 A-fragments x 2 B-fragments -> four
// v_wmma_f32_16x16x32_f16 per K-step (8 b128 loads per 4 WMMAs).
// The K-loop is software-pipelined: fragments for step kk+32 are loaded
// before the WMMAs consuming step kk, so s_wait covers a full iteration.
// Edge tiles: pointers are clamped in-bounds and the extra WMMAs computed
// unconditionally (harmless); only the stores are guarded -> no exec-mask
// manipulation in the hot loop, EXEC stays all-ones for WMMA.
// A/B fragments: contiguous k = 16*(lane>>4) + s in both; C/D layout per CDNA5 ISA 7.12.2.
// act: 0 = none, 1 = relu, 2 = leaky_relu(0.01)
// =====================================================================
__global__ __launch_bounds__(256)
void gemm_wmma_f16(const _Float16* __restrict__ A, const _Float16* __restrict__ W,
                   const float* __restrict__ bias, float* __restrict__ C,
                   int M, int N, int K, int act)
{
    const int lane = threadIdx.x & 31;
    const int wave = threadIdx.x >> 5;
    const int m0 = (blockIdx.y * 2 + (wave >> 2)) * 32;   // two 16-row tiles
    const int n0 = (blockIdx.x * 4 + (wave & 3)) * 32;    // two 16-col tiles
    if (m0 >= M || n0 >= N) return;                        // wave-uniform
    const bool hasM1 = (m0 + 32) <= M;
    const bool hasN1 = (n0 + 32) <= N;

    const int r  = lane & 15;
    const int hi = lane >> 4;

    const _Float16* __restrict__ a0p = A + (size_t)(m0 + r) * K + hi * 16;
    const _Float16* __restrict__ a1p = A + (size_t)((hasM1 ? m0 + 16 : m0) + r) * K + hi * 16;
    const _Float16* __restrict__ b0p = W + (size_t)(n0 + r) * K + hi * 16;
    const _Float16* __restrict__ b1p = W + (size_t)((hasN1 ? n0 + 16 : n0) + r) * K + hi * 16;

    v8f c00 = {}, c01 = {}, c10 = {}, c11 = {};
    v16h a0, a1, b0, b1;

#define LOAD_FRAGS(kk, A0, A1, B0, B1)                                    \
    {                                                                     \
        v8h l0 = *(const v8h*)(a0p + (kk));                               \
        v8h h0 = *(const v8h*)(a0p + (kk) + 8);                           \
        v8h l1 = *(const v8h*)(a1p + (kk));                               \
        v8h h1 = *(const v8h*)(a1p + (kk) + 8);                           \
        _Pragma("unroll")                                                 \
        for (int j = 0; j < 8; ++j) {                                     \
            A0[j] = l0[j]; A0[8 + j] = h0[j];                             \
            A1[j] = l1[j]; A1[8 + j] = h1[j];                             \
        }                                                                 \
        B0 = *(const v16h*)(b0p + (kk));                                  \
        B1 = *(const v16h*)(b1p + (kk));                                  \
    }

    LOAD_FRAGS(0, a0, a1, b0, b1)
    for (int kk = 32; kk < K; kk += 32) {
        if (kk + 32 < K) {   // prefetch one step past the loads issued now
            __builtin_prefetch(a0p + kk + 32, 0, 3);
            __builtin_prefetch(a1p + kk + 32, 0, 3);
            __builtin_prefetch(b0p + kk + 32, 0, 3);
            __builtin_prefetch(b1p + kk + 32, 0, 3);
        }
        v16h na0, na1, nb0, nb1;
        LOAD_FRAGS(kk, na0, na1, nb0, nb1)
        c00 = __builtin_amdgcn_wmma_f32_16x16x32_f16(false, a0, false, b0, (short)0, c00, false, false);
        c01 = __builtin_amdgcn_wmma_f32_16x16x32_f16(false, a0, false, b1, (short)0, c01, false, false);
        c10 = __builtin_amdgcn_wmma_f32_16x16x32_f16(false, a1, false, b0, (short)0, c10, false, false);
        c11 = __builtin_amdgcn_wmma_f32_16x16x32_f16(false, a1, false, b1, (short)0, c11, false, false);
        a0 = na0; a1 = na1; b0 = nb0; b1 = nb1;
    }
    c00 = __builtin_amdgcn_wmma_f32_16x16x32_f16(false, a0, false, b0, (short)0, c00, false, false);
    c01 = __builtin_amdgcn_wmma_f32_16x16x32_f16(false, a0, false, b1, (short)0, c01, false, false);
    c10 = __builtin_amdgcn_wmma_f32_16x16x32_f16(false, a1, false, b0, (short)0, c10, false, false);
    c11 = __builtin_amdgcn_wmma_f32_16x16x32_f16(false, a1, false, b1, (short)0, c11, false, false);
#undef LOAD_FRAGS

#define STORE_ELEM(val, row, col)                                         \
    { float v = (val); if (bias) v += bias[col];                          \
      if (act == 1)      v = fmaxf(v, 0.0f);                              \
      else if (act == 2) v = (v > 0.0f) ? v : 0.01f * v;                  \
      C[(size_t)(row) * N + (col)] = v; }

    // C/D layout: VGPR i -> M = i (lanes 0-15) or i+8 (lanes 16-31), N = lane&15
#pragma unroll
    for (int i = 0; i < 8; ++i) {
        const int row0 = m0 + i + hi * 8;
        const int row1 = row0 + 16;
        const int col0 = n0 + r;
        const int col1 = n0 + 16 + r;
        STORE_ELEM(c00[i], row0, col0)
        if (hasN1)          STORE_ELEM(c01[i], row0, col1)
        if (hasM1)          STORE_ELEM(c10[i], row1, col0)
        if (hasM1 && hasN1) STORE_ELEM(c11[i], row1, col1)
    }
#undef STORE_ELEM
}

// =====================================================================
// value MLP stage 1: v1[bl,d] = relu(min(values[bl],512)*ve_w1[d] + ve_b1[d])
// =====================================================================
__global__ void value_mlp1_kernel(const float* __restrict__ values,
                                  const float* __restrict__ w1,
                                  const float* __restrict__ b1,
                                  float* __restrict__ out)
{
    int idx = blockIdx.x * blockDim.x + threadIdx.x;
    if (idx >= BLROWS * DMODEL) return;
    int bl = idx / DMODEL, d = idx % DMODEL;
    float x = fminf(values[bl], MAXVAL);
    out[idx] = fmaxf(x * w1[d] + b1[d], 0.0f);
}

// =====================================================================
// h = ( LN(embed[src]; ge_g, ge_b) + LN(v2; ve_g, ve_b) ) * (!mask)
// =====================================================================
__global__ __launch_bounds__(256)
void embed_combine_kernel(const int* __restrict__ src,
                          const float* __restrict__ embed,
                          const float* __restrict__ geg, const float* __restrict__ geb,
                          const float* __restrict__ v2,
                          const float* __restrict__ veg, const float* __restrict__ veb,
                          const unsigned char* __restrict__ mask,
                          float* __restrict__ h)
{
    const int row = blockIdx.x;
    const int tid = threadIdx.x;
    __shared__ float e[DMODEL], vv[DMODEL];
    __shared__ float rs[256], rss[256];

    const int tok = src[row];
    const float* er = embed + (size_t)tok * DMODEL;
    const float* vr = v2 + (size_t)row * DMODEL;

    float s1 = 0.f, q1 = 0.f, s2 = 0.f, q2 = 0.f;
    for (int c = tid; c < DMODEL; c += 256) {
        float a = er[c]; e[c]  = a; s1 += a; q1 += a * a;
        float b = vr[c]; vv[c] = b; s2 += b; q2 += b * b;
    }
    rs[tid] = s1; rss[tid] = q1; __syncthreads();
    for (int k = 128; k > 0; k >>= 1) { if (tid < k) { rs[tid] += rs[tid+k]; rss[tid] += rss[tid+k]; } __syncthreads(); }
    const float m1 = rs[0] * (1.f / DMODEL);
    const float i1 = rsqrtf(rss[0] * (1.f / DMODEL) - m1 * m1 + LN_EPS);
    __syncthreads();
    rs[tid] = s2; rss[tid] = q2; __syncthreads();
    for (int k = 128; k > 0; k >>= 1) { if (tid < k) { rs[tid] += rs[tid+k]; rss[tid] += rss[tid+k]; } __syncthreads(); }
    const float m2 = rs[0] * (1.f / DMODEL);
    const float i2 = rsqrtf(rss[0] * (1.f / DMODEL) - m2 * m2 + LN_EPS);

    const float keep = mask[row] ? 0.0f : 1.0f;
    for (int c = tid; c < DMODEL; c += 256) {
        float a = (e[c]  - m1) * i1 * geg[c] + geb[c];
        float b = (vv[c] - m2) * i2 * veg[c] + veb[c];
        h[(size_t)row * DMODEL + c] = (a + b) * keep;
    }
}

// =====================================================================
// generic LayerNorm over rows (g/b optional), one block per row
// =====================================================================
__global__ __launch_bounds__(256)
void ln_kernel(const float* __restrict__ in, float* __restrict__ out,
               const float* __restrict__ g, const float* __restrict__ b, int cols)
{
    const int row = blockIdx.x;
    const int tid = threadIdx.x;
    const float* x = in + (size_t)row * cols;
    __shared__ float rs[256], rss[256];
    float s = 0.f, q = 0.f;
    for (int c = tid; c < cols; c += 256) { float v = x[c]; s += v; q += v * v; }
    rs[tid] = s; rss[tid] = q; __syncthreads();
    for (int k = 128; k > 0; k >>= 1) { if (tid < k) { rs[tid] += rs[tid+k]; rss[tid] += rss[tid+k]; } __syncthreads(); }
    const float m   = rs[0] / cols;
    const float inv = rsqrtf(rss[0] / cols - m * m + LN_EPS);
    for (int c = tid; c < cols; c += 256) {
        float v = (x[c] - m) * inv;
        if (g) v = v * g[c] + b[c];
        out[(size_t)row * cols + c] = v;
    }
}

// h = LN( (a + b) / 2 ), cols = DMODEL
__global__ __launch_bounds__(256)
void avg_ln_kernel(const float* __restrict__ A, const float* __restrict__ B,
                   float* __restrict__ out)
{
    const int row = blockIdx.x;
    const int tid = threadIdx.x;
    __shared__ float buf[DMODEL];
    __shared__ float rs[256], rss[256];
    float s = 0.f, q = 0.f;
    for (int c = tid; c < DMODEL; c += 256) {
        float v = 0.5f * (A[(size_t)row * DMODEL + c] + B[(size_t)row * DMODEL + c]);
        buf[c] = v; s += v; q += v * v;
    }
    rs[tid] = s; rss[tid] = q; __syncthreads();
    for (int k = 128; k > 0; k >>= 1) { if (tid < k) { rs[tid] += rs[tid+k]; rss[tid] += rss[tid+k]; } __syncthreads(); }
    const float m   = rs[0] * (1.f / DMODEL);
    const float inv = rsqrtf(rss[0] * (1.f / DMODEL) - m * m + LN_EPS);
    for (int c = tid; c < DMODEL; c += 256)
        out[(size_t)row * DMODEL + c] = (buf[c] - m) * inv;
}

// =====================================================================
// sequence lengths (count of !mask) and smart flip (row gather)
// =====================================================================
__global__ void lengths_kernel(const unsigned char* __restrict__ mask, int* __restrict__ lens)
{
    const int b = blockIdx.x, tid = threadIdx.x;
    __shared__ int rs[256];
    int s = 0;
    for (int l = tid; l < LSEQ; l += 256) s += (mask[b * LSEQ + l] == 0) ? 1 : 0;
    rs[tid] = s; __syncthreads();
    for (int k = 128; k > 0; k >>= 1) { if (tid < k) rs[tid] += rs[tid+k]; __syncthreads(); }
    if (tid == 0) lens[b] = rs[0];
}

__global__ void flip_kernel(const float* __restrict__ in, float* __restrict__ out,
                            const int* __restrict__ lens)
{
    const int l = blockIdx.x, b = blockIdx.y, tid = threadIdx.x;
    const int len = lens[b];
    const int pos = (l < len) ? (len - 1 - l) : l;
    const float* s = in  + ((size_t)b * LSEQ + pos) * DMODEL;
    float*       d = out + ((size_t)b * LSEQ + l)   * DMODEL;
    d[tid] = s[tid];
    d[tid + 256] = s[tid + 256];
}

// =====================================================================
// dt = softplus(zx[..., 2176+h] + dt_bias[h]);  dA = exp(-exp(A_log[h]) * dt)
// =====================================================================
__global__ void dtda_kernel(const float* __restrict__ zx,
                            const float* __restrict__ dt_bias,
                            const float* __restrict__ A_log,
                            float* __restrict__ dtb, float* __restrict__ dab)
{
    int idx = blockIdx.x * blockDim.x + threadIdx.x;
    if (idx >= BLROWS * NHEADS) return;
    int hh = idx & (NHEADS - 1);
    int bl = idx >> 4;
    float dt = softplusf(zx[(size_t)bl * DINPROJ + (DINNER + CONVDIM) + hh] + dt_bias[hh]);
    dtb[idx] = dt;
    dab[idx] = expf(-expf(A_log[hh]) * dt);
}

// =====================================================================
// causal depthwise conv(4) over xBC (= zx cols [1024, 2176)) + SiLU
// =====================================================================
__global__ void conv_kernel(const float* __restrict__ zx,
                            const float* __restrict__ cw,   // [CONVDIM,4]
                            const float* __restrict__ cb,   // [CONVDIM]
                            float* __restrict__ out)        // [B,L,CONVDIM]
{
    int idx = blockIdx.x * blockDim.x + threadIdx.x;
    if (idx >= BLROWS * CONVDIM) return;
    const int c = idx % CONVDIM;
    const int l = (idx / CONVDIM) % LSEQ;
    const int b = idx / (CONVDIM * LSEQ);
    float acc = cb[c];
#pragma unroll
    for (int j = 0; j < DCONV; ++j) {
        int ls = l + j - (DCONV - 1);
        if (ls >= 0)
            acc += zx[((size_t)b * LSEQ + ls) * DINPROJ + DINNER + c] * cw[c * DCONV + j];
    }
    out[idx] = siluf(acc);
}

// =====================================================================
// selective-state scan: one block per (b, head); 64x64 f32 state in regs.
// thread t owns state row p = t>>2, n-slice [(t&3)*16, +16).
// =====================================================================
#define TCH 8
__global__ __launch_bounds__(256)
void scan_kernel(const float* __restrict__ conv,   // [B,L,CONVDIM]
                 const float* __restrict__ dtb,    // [B*L,NHEADS]
                 const float* __restrict__ dab,
                 const float* __restrict__ Dskip,  // [NHEADS]
                 float* __restrict__ y)            // [B,L,DINNER]
{
    const int b  = blockIdx.x / NHEADS;
    const int hh = blockIdx.x % NHEADS;
    const int tid = threadIdx.x;
    const int p  = tid >> 2;
    const int nq = (tid & 3) * 16;
    const float Dh = Dskip[hh];

    float st[16];
#pragma unroll
    for (int j = 0; j < 16; ++j) st[j] = 0.0f;

    __shared__ float sbuf[TCH][192];   // per step: x[64] | B[64] | C[64]

    for (int l0 = 0; l0 < LSEQ; l0 += TCH) {
        for (int idx = tid; idx < TCH * 192; idx += 256) {
            const int t = idx / 192, c = idx % 192;
            const size_t base = ((size_t)b * LSEQ + (l0 + t)) * CONVDIM;
            float v;
            if (c < 64)       v = conv[base + hh * HEADDIM + c];            // x
            else if (c < 128) v = conv[base + DINNER + (c - 64)];           // B
            else              v = conv[base + DINNER + DSTATE + (c - 128)]; // C
            sbuf[t][c] = v;
        }
        __syncthreads();
#pragma unroll
        for (int t = 0; t < TCH; ++t) {
            const int l = l0 + t;
            const float dt = dtb[((size_t)b * LSEQ + l) * NHEADS + hh];
            const float dA = dab[((size_t)b * LSEQ + l) * NHEADS + hh];
            const float xp = sbuf[t][p];
            const float coef = dt * xp;
            float acc = 0.0f;
#pragma unroll
            for (int j = 0; j < 16; ++j) {
                st[j] = st[j] * dA + coef * sbuf[t][64 + nq + j];
                acc  += st[j] * sbuf[t][128 + nq + j];
            }
            acc += __shfl_xor(acc, 1, 32);
            acc += __shfl_xor(acc, 2, 32);
            if ((tid & 3) == 0)
                y[((size_t)b * LSEQ + l) * DINNER + hh * HEADDIM + p] = acc + Dh * xp;
        }
        __syncthreads();
    }
}

// =====================================================================
// y = rmsnorm( y * silu(z) ) * norm_w   (in place; z = zx cols [0,1024))
// =====================================================================
__global__ __launch_bounds__(256)
void gatenorm_kernel(float* __restrict__ y, const float* __restrict__ zx,
                     const float* __restrict__ nw)
{
    const int row = blockIdx.x, tid = threadIdx.x;
    float* yr = y + (size_t)row * DINNER;
    const float* zr = zx + (size_t)row * DINPROJ;
    __shared__ float rs[256];
    float g4[4]; float ss = 0.f;
#pragma unroll
    for (int j = 0; j < 4; ++j) {
        const int c = tid + j * 256;
        float gv = yr[c] * siluf(zr[c]);
        g4[j] = gv; ss += gv * gv;
    }
    rs[tid] = ss; __syncthreads();
    for (int k = 128; k > 0; k >>= 1) { if (tid < k) rs[tid] += rs[tid+k]; __syncthreads(); }
    const float inv = rsqrtf(rs[0] * (1.f / DINNER) + LN_EPS);
#pragma unroll
    for (int j = 0; j < 4; ++j) {
        const int c = tid + j * 256;
        yr[c] = g4[j] * inv * nw[c];
    }
}

// =====================================================================
// mlm[row] = dot(e2[row,:], w3) + b3   (one block per row)
// =====================================================================
__global__ __launch_bounds__(256)
void mlm_kernel(const float* __restrict__ e2, const float* __restrict__ w3,
                const float* __restrict__ b3, float* __restrict__ out)
{
    const int row = blockIdx.x, tid = threadIdx.x;
    const float* x = e2 + (size_t)row * DMODEL;
    __shared__ float rs[256];
    float s = x[tid] * w3[tid] + x[tid + 256] * w3[tid + 256];
    rs[tid] = s; __syncthreads();
    for (int k = 128; k > 0; k >>= 1) { if (tid < k) rs[tid] += rs[tid+k]; __syncthreads(); }
    if (tid == 0) out[row] = rs[0] + b3[0];
}

__global__ void cell_copy_kernel(const float* __restrict__ h, float* __restrict__ out)
{
    int idx = blockIdx.x * blockDim.x + threadIdx.x;
    if (idx >= NB * DMODEL) return;
    int b = idx / DMODEL, d = idx % DMODEL;
    out[idx] = h[(size_t)b * LSEQ * DMODEL + d];
}

// =====================================================================
// tiny matvec for the classifier head: out[b,n] = act(dot(in[b], W[n]) + bias[n])
// =====================================================================
__global__ __launch_bounds__(128)
void matvec_kernel(const float* __restrict__ in, int in_stride,
                   const float* __restrict__ W, const float* __restrict__ bias,
                   float* __restrict__ out, int K, int N, int act)
{
    const int n = blockIdx.x, b = blockIdx.y, tid = threadIdx.x;
    const float* x = in + (size_t)b * in_stride;
    const float* w = W + (size_t)n * K;
    __shared__ float rs[128];
    float s = 0.f;
    for (int c = tid; c < K; c += 128) s += x[c] * w[c];
    rs[tid] = s; __syncthreads();
    for (int k = 64; k > 0; k >>= 1) { if (tid < k) rs[tid] += rs[tid+k]; __syncthreads(); }
    if (tid == 0) {
        float v = rs[0] + (bias ? bias[n] : 0.0f);
        if (act == 1) v = fmaxf(v, 0.0f);
        out[(size_t)b * N + n] = v;
    }
}

// =====================================================================
// host-side helpers
// =====================================================================
static inline void cvt16(const float* in, _Float16* out, int n, hipStream_t st)
{
    f32_to_f16_kernel<<<(n + 1023) / 1024, 256, 0, st>>>(in, out, n);
}

static inline void launch_gemm(const float* A, _Float16* a16, const _Float16* W16,
                               const float* bias, float* C,
                               int M, int N, int K, int act, hipStream_t st)
{
    cvt16(A, a16, M * K, st);
    const int pairsN = (N / 16 + 1) / 2;
    const int pairsM = (M / 16 + 1) / 2;
    dim3 grid((pairsN + 3) / 4, (pairsM + 1) / 2);
    gemm_wmma_f16<<<grid, 256, 0, st>>>(a16, W16, bias, C, M, N, K, act);
}

static void run_mamba(const float* u, const _Float16* in_w16, const float* cw, const float* cb,
                      const float* alog, const float* dtbias, const float* dsk,
                      const float* nw, const _Float16* ow16, float* out,
                      float* zx, float* conv, float* dtb, float* dab, float* ybuf,
                      _Float16* a16, hipStream_t st)
{
    launch_gemm(u, a16, in_w16, nullptr, zx, BLROWS, DINPROJ, DMODEL, 0, st);
    dtda_kernel<<<(BLROWS * NHEADS + 255) / 256, 256, 0, st>>>(zx, dtbias, alog, dtb, dab);
    conv_kernel<<<(BLROWS * CONVDIM + 255) / 256, 256, 0, st>>>(zx, cw, cb, conv);
    scan_kernel<<<NB * NHEADS, 256, 0, st>>>(conv, dtb, dab, dsk, ybuf);
    gatenorm_kernel<<<BLROWS, 256, 0, st>>>(ybuf, zx, nw);
    launch_gemm(ybuf, a16, ow16, nullptr, out, BLROWS, DMODEL, DINNER, 0, st);
}

extern "C" void kernel_launch(void* const* d_in, const int* in_sizes, int n_in,
                              void* d_out, int out_size, void* d_ws, size_t ws_size,
                              hipStream_t stream)
{
    // ---- inputs in setup_inputs() order ----
    const int*           src       = (const int*)d_in[0];
    const float*         values    = (const float*)d_in[1];
    const unsigned char* mask      = (const unsigned char*)d_in[2];
    const float* embed    = (const float*)d_in[3];
    const float* ge_ln_g  = (const float*)d_in[4];
    const float* ge_ln_b  = (const float*)d_in[5];
    const float* ve_w1    = (const float*)d_in[6];
    const float* ve_b1    = (const float*)d_in[7];
    const float* ve_w2    = (const float*)d_in[8];
    const float* ve_b2    = (const float*)d_in[9];
    const float* ve_ln_g  = (const float*)d_in[10];
    const float* ve_ln_b  = (const float*)d_in[11];
    const float* in_proj_w = (const float*)d_in[12];  // [2, 2192, 512]
    const float* conv_w_f = (const float*)d_in[13];
    const float* conv_b_f = (const float*)d_in[14];
    const float* A_log_f  = (const float*)d_in[15];
    const float* dt_bias_f= (const float*)d_in[16];
    const float* D_f      = (const float*)d_in[17];
    const float* norm_w_f = (const float*)d_in[18];
    const float* conv_w_r = (const float*)d_in[19];
    const float* conv_b_r = (const float*)d_in[20];
    const float* A_log_r  = (const float*)d_in[21];
    const float* dt_bias_r= (const float*)d_in[22];
    const float* D_r      = (const float*)d_in[23];
    const float* norm_w_r = (const float*)d_in[24];
    const float* out_proj_w = (const float*)d_in[25]; // [2, 512, 1024]
    const float* expr_w1  = (const float*)d_in[26];
    const float* expr_b1  = (const float*)d_in[27];
    const float* expr_w2  = (const float*)d_in[28];
    const float* expr_b2  = (const float*)d_in[29];
    const float* expr_w3  = (const float*)d_in[30];
    const float* expr_b3  = (const float*)d_in[31];
    const float* cls_w1   = (const float*)d_in[32];
    const float* cls_b1   = (const float*)d_in[33];
    const float* cls_ln1_g= (const float*)d_in[34];
    const float* cls_ln1_b= (const float*)d_in[35];
    const float* cls_w2   = (const float*)d_in[36];
    const float* cls_b2   = (const float*)d_in[37];
    const float* cls_ln2_g= (const float*)d_in[38];
    const float* cls_ln2_b= (const float*)d_in[39];
    const float* cls_wo   = (const float*)d_in[40];
    const float* cls_bo   = (const float*)d_in[41];
    (void)in_sizes; (void)n_in; (void)out_size; (void)ws_size;

    float* outp = (float*)d_out;   // [mlm 4096 | cell 2048 | cls 656]

    // ---- workspace layout ----
    constexpr size_t SZ_H  = (size_t)BLROWS * DMODEL;   // 2,097,152 f32
    constexpr size_t SZ_ZX = (size_t)BLROWS * DINPROJ;  // 8,978,432 f32
    constexpr size_t SZ_CV = (size_t)BLROWS * CONVDIM;  // 4,718,592 f32
    constexpr size_t SZ_DT = (size_t)BLROWS * NHEADS;
    constexpr size_t SZ_Y  = (size_t)BLROWS * DINNER;   // 4,194,304 f32
    constexpr size_t W16_VE  = (size_t)DMODEL * DMODEL;        // 262,144
    constexpr size_t W16_IN  = (size_t)2 * DINPROJ * DMODEL;   // 2,244,608
    constexpr size_t W16_OUT = (size_t)2 * DMODEL * DINNER;    // 1,048,576
    constexpr size_t W16_E   = (size_t)DMODEL * DMODEL;

    float* ws   = (float*)d_ws;
    float* h    = ws;
    float* hn   = h    + SZ_H;
    float* rev  = hn   + SZ_H;
    float* outf = rev  + SZ_H;
    float* outr = outf + SZ_H;
    float* zx   = outr + SZ_H;
    float* conv = zx   + SZ_ZX;
    float* dtb  = conv + SZ_CV;
    float* dab  = dtb  + SZ_DT;
    float* ybuf = dab  + SZ_DT;
    int*   lens = (int*)(ybuf + SZ_Y);
    float* cb1  = (float*)(lens + 16);
    float* cb2  = cb1 + 2048;
    // f16 staging region (offset is a multiple of 32B: 4112 floats past ybuf end)
    _Float16* w16_ve  = (_Float16*)(cb2 + 2048);
    _Float16* w16_in  = w16_ve  + W16_VE;
    _Float16* w16_out = w16_in  + W16_IN;
    _Float16* w16_e1  = w16_out + W16_OUT;
    _Float16* w16_e2  = w16_e1  + W16_E;
    _Float16* a16     = w16_e2  + W16_E;     // BLROWS * DINNER halves (max M*K)
    // scratch aliased onto zx (disjoint lifetimes with the mamba intermediates)
    float* s1 = zx;
    float* s2 = zx + SZ_H;

    // ---- one-time (per launch) weight conversion to f16 ----
    cvt16(ve_w2,     w16_ve,  (int)W16_VE,  stream);
    cvt16(in_proj_w, w16_in,  (int)W16_IN,  stream);
    cvt16(out_proj_w,w16_out, (int)W16_OUT, stream);
    cvt16(expr_w1,   w16_e1,  (int)W16_E,   stream);
    cvt16(expr_w2,   w16_e2,  (int)W16_E,   stream);

    // ---- embedding + value MLP ----
    value_mlp1_kernel<<<(BLROWS * DMODEL + 255) / 256, 256, 0, stream>>>(values, ve_w1, ve_b1, s1);
    launch_gemm(s1, a16, w16_ve, ve_b2, s2, BLROWS, DMODEL, DMODEL, 0, stream);
    embed_combine_kernel<<<BLROWS, 256, 0, stream>>>(src, embed, ge_ln_g, ge_ln_b,
                                                     s2, ve_ln_g, ve_ln_b, mask, h);
    lengths_kernel<<<NB, 256, 0, stream>>>(mask, lens);

    // ---- bidirectional Mamba2 layers ----
    for (int i = 0; i < 2; ++i) {
        const _Float16* inw16 = w16_in  + (size_t)i * DINPROJ * DMODEL;
        const _Float16* ow16  = w16_out + (size_t)i * DMODEL * DINNER;

        ln_kernel<<<BLROWS, 256, 0, stream>>>(h, hn, nullptr, nullptr, DMODEL);

        run_mamba(hn, inw16, conv_w_f + (size_t)i * CONVDIM * DCONV, conv_b_f + (size_t)i * CONVDIM,
                  A_log_f + i * NHEADS, dt_bias_f + i * NHEADS, D_f + i * NHEADS,
                  norm_w_f + (size_t)i * DINNER, ow16, outf,
                  zx, conv, dtb, dab, ybuf, a16, stream);

        flip_kernel<<<dim3(LSEQ, NB), 256, 0, stream>>>(hn, rev, lens);

        run_mamba(rev, inw16, conv_w_r + (size_t)i * CONVDIM * DCONV, conv_b_r + (size_t)i * CONVDIM,
                  A_log_r + i * NHEADS, dt_bias_r + i * NHEADS, D_r + i * NHEADS,
                  norm_w_r + (size_t)i * DINNER, ow16, outr,
                  zx, conv, dtb, dab, ybuf, a16, stream);

        flip_kernel<<<dim3(LSEQ, NB), 256, 0, stream>>>(outr, rev, lens);
        avg_ln_kernel<<<BLROWS, 256, 0, stream>>>(outf, rev, h);
    }

    // ---- expression head (mlm) ----
    launch_gemm(h,  a16, w16_e1, expr_b1, s1, BLROWS, DMODEL, DMODEL, 2, stream);
    launch_gemm(s1, a16, w16_e2, expr_b2, s2, BLROWS, DMODEL, DMODEL, 2, stream);
    mlm_kernel<<<BLROWS, 256, 0, stream>>>(s2, expr_w3, expr_b3, outp);

    // ---- cell + classifier head ----
    float* cellp = outp + BLROWS;              // [4, 512]
    float* clsp  = cellp + NB * DMODEL;        // [4, 164]
    cell_copy_kernel<<<(NB * DMODEL + 255) / 256, 256, 0, stream>>>(h, cellp);
    matvec_kernel<<<dim3(DMODEL, NB), 128, 0, stream>>>(cellp, DMODEL, cls_w1, cls_b1, cb1, DMODEL, DMODEL, 1);
    ln_kernel<<<NB, 256, 0, stream>>>(cb1, cb2, cls_ln1_g, cls_ln1_b, DMODEL);
    matvec_kernel<<<dim3(DMODEL, NB), 128, 0, stream>>>(cb2, DMODEL, cls_w2, cls_b2, cb1, DMODEL, DMODEL, 1);
    ln_kernel<<<NB, 256, 0, stream>>>(cb1, cb2, cls_ln2_g, cls_ln2_b, DMODEL);
    matvec_kernel<<<dim3(NCLSOUT, NB), 128, 0, stream>>>(cb2, DMODEL, cls_wo, cls_bo, clsp, DMODEL, NCLSOUT, 0);
}
